// TransformerReadout_2491081031691
// MI455X (gfx1250) — matrix operand order repaired and checked by reading, not compiled, not run on hardware
//
#include <hip/hip_runtime.h>
#include <hip/hip_bf16.h>

// ---------------------------------------------------------------------------
// TransformerReadout, CDNA5 (gfx1250) wave32 + WMMA bf16 implementation.
// Only the cls row (s==0) feeds the output, so Q/attention/FFN are computed
// for the cls row only. Dominant cost: K/V projection GEMM (M=65536, N=1024,
// K=512) with v_wmma_f32_16x16x32_bf16, fp32 x converted to bf16 on the fly.
// Pipelined: global loads for tile kt+1 are staged in VGPRs during the WMMAs
// of tile kt; bf16 outputs are staged through LDS for b128 coalesced stores.
// ---------------------------------------------------------------------------

#define D_   512
#define H_   8
#define DH_  64
#define FF_  2048
#define OUT_ 128
#define G_   256
#define N_   255
#define S_   256   // N_ + 1 (cls)

typedef __attribute__((ext_vector_type(16))) __bf16 v16bf;
typedef __attribute__((ext_vector_type(8)))  __bf16 v8bf;
typedef __attribute__((ext_vector_type(8)))  float  v8f;

__device__ __forceinline__ unsigned pack2_bf16(float a, float b) {
    unsigned short ua = __builtin_bit_cast(unsigned short, (__bf16)a);
    unsigned short ub = __builtin_bit_cast(unsigned short, (__bf16)b);
    return (unsigned)ua | ((unsigned)ub << 16);
}

// LDS tile geometry: 128 rows x 64 k, padded row stride 72 elems (144 B)
// -> 16 consecutive rows map to distinct bank groups (36*r mod 64 distinct).
#define SSTR 72

// A fragment (16x32 bf16), CDNA5 ISA 7.12.2:
// lane l: row = frag_row + (l&15), half = l>>4
//   elements 0..7  = K half*8 .. half*8+7
//   elements 8..15 = K half*8+16 .. half*8+23
__device__ __forceinline__ v16bf load_fragA(const __bf16* s, int row, int half,
                                            int ks) {
    const __bf16* p = s + row * SSTR + ks * 32 + half * 8;
    v8bf lo = *reinterpret_cast<const v8bf*>(p);
    v8bf hi = *reinterpret_cast<const v8bf*>(p + 16);
    return __builtin_shufflevector(lo, hi, 0,1,2,3,4,5,6,7,8,9,10,11,12,13,14,15);
}

// B fragment (32x16 bf16, stored as W rows [n][k]):
// lane l: n = frag_n + (l&15), half = l>>4; elements 0..7 = K half*8..+7,
//   elements 8..15 = K half*8+16..+23.
__device__ __forceinline__ v16bf load_fragB(const __bf16* s, int nrow, int half,
                                            int ks) {
    const __bf16* p = s + nrow * SSTR + ks * 32 + half * 8;
    v8bf lo = *reinterpret_cast<const v8bf*>(p);
    v8bf hi = *reinterpret_cast<const v8bf*>(p + 16);
    return __builtin_shufflevector(lo, hi, 0,1,2,3,4,5,6,7,8,9,10,11,12,13,14,15);
}

// ---------------- generic WMMA GEMM:  C = A @ B^T + bias (+epilogue) --------
// MODE 0: A = seq (fp32 gather from x/cls, M=G*S), epilogue: bf16 store  (KV)
// MODE 1: A = bf16, epilogue: relu -> bf16 store                       (FFN1)
// MODE 2: A = bf16, epilogue: f32 store + bias + residual              (FFN2)
// MODE 3: A = bf16, epilogue: f32 store + bias                         (OUT)
struct GemmArgs {
    const float*  xsrc;
    const float*  cls;
    const __bf16* Abf;    // lda = K
    const __bf16* Bbf;    // W rows, [Nrows][K]
    const float*  bias;   // [N]
    const float*  resid;  // MODE 2, [M][512]
    __bf16*       Cbf;
    float*        Cf;
    int           K;      // multiple of 64
    int           ldc;
};

template <int MODE>
__global__ __launch_bounds__(256) void gemm_wmma(GemmArgs args) {
    __shared__ __bf16 smem[2 * 128 * SSTR];   // 36 KB: sA | sB
    __bf16* sA = smem;
    __bf16* sB = smem + 128 * SSTR;

    const int tid  = threadIdx.x;
    const int lane = tid & 31;
    const int wid  = tid >> 5;
    const int wm   = wid & 1;        // 0..1 -> 64 rows
    const int wn   = wid >> 1;       // 0..3 -> 32 cols
    const int lrow = lane & 15;
    const int half = lane >> 4;

    const int m0 = blockIdx.y * 128;
    const int n0 = blockIdx.x * 128;

    v8f acc[4][2] = {};

    // staging registers for the next k-tile (global -> regs -> LDS pipeline)
    uint4  breg[4];
    uint4  areg[4];      // MODE != 0
    float4 freg[8];      // MODE == 0

    auto load_tiles = [&](int kt) {
        const int kb = kt << 6;
#pragma unroll
        for (int it = 0; it < 4; ++it) {
            int c   = tid + it * 256;
            int row = c >> 3;
            int kc  = (c & 7) << 3;
            breg[it] = *reinterpret_cast<const uint4*>(
                args.Bbf + (size_t)(n0 + row) * args.K + kb + kc);
        }
        if (MODE == 0) {
#pragma unroll
            for (int it = 0; it < 8; ++it) {
                int c  = tid + it * 256;
                int r  = c >> 4;
                int cc = (c & 15) << 2;
                int m  = m0 + r;
                int g  = m >> 8;
                int s  = m & 255;
                // x row index = g*255 + (s-1) = m - g - 1
                const float* sp = (s == 0)
                    ? (args.cls + kb + cc)
                    : (args.xsrc + (size_t)(m - g - 1) * D_ + kb + cc);
                freg[it] = *reinterpret_cast<const float4*>(sp);
            }
        } else {
#pragma unroll
            for (int it = 0; it < 4; ++it) {
                int c   = tid + it * 256;
                int row = c >> 3;
                int kc  = (c & 7) << 3;
                areg[it] = *reinterpret_cast<const uint4*>(
                    args.Abf + (size_t)(m0 + row) * args.K + kb + kc);
            }
        }
    };

    auto store_tiles = [&]() {
#pragma unroll
        for (int it = 0; it < 4; ++it) {
            int c   = tid + it * 256;
            int row = c >> 3;
            int kc  = (c & 7) << 3;
            *reinterpret_cast<uint4*>(&sB[row * SSTR + kc]) = breg[it];
        }
        if (MODE == 0) {
#pragma unroll
            for (int it = 0; it < 8; ++it) {
                int c  = tid + it * 256;
                int r  = c >> 4;
                int cc = (c & 15) << 2;
                uint2 u;
                u.x = pack2_bf16(freg[it].x, freg[it].y);
                u.y = pack2_bf16(freg[it].z, freg[it].w);
                *reinterpret_cast<uint2*>(&sA[r * SSTR + cc]) = u;
            }
        } else {
#pragma unroll
            for (int it = 0; it < 4; ++it) {
                int c   = tid + it * 256;
                int row = c >> 3;
                int kc  = (c & 7) << 3;
                *reinterpret_cast<uint4*>(&sA[row * SSTR + kc]) = areg[it];
            }
        }
    };

    const int nk = args.K >> 6;
    load_tiles(0);
    for (int kt = 0; kt < nk; ++kt) {
        __syncthreads();              // previous compute done reading LDS
        store_tiles();
        __syncthreads();
        if (kt + 1 < nk) load_tiles(kt + 1);   // overlap with WMMAs below

#pragma unroll
        for (int ks = 0; ks < 2; ++ks) {
            v16bf af[4], bfr[2];
#pragma unroll
            for (int fa = 0; fa < 4; ++fa)
                af[fa] = load_fragA(sA, wm * 64 + fa * 16 + lrow, half, ks);
#pragma unroll
            for (int fb = 0; fb < 2; ++fb)
                bfr[fb] = load_fragB(sB, wn * 32 + fb * 16 + lrow, half, ks);
#pragma unroll
            for (int fa = 0; fa < 4; ++fa)
#pragma unroll
                for (int fb = 0; fb < 2; ++fb)
                    acc[fa][fb] = __builtin_amdgcn_wmma_f32_16x16x32_bf16(
                        false, af[fa], false, bfr[fb], (short)0, acc[fa][fb],
                        false, false);
        }
    }

    // ---- epilogue. C layout: lane l: n = l&15, elem i: m = i + 8*(l>>4) ----
    if (MODE == 0 || MODE == 1) {
        // stage bf16 tile in LDS, then b128 coalesced global stores
        __syncthreads();
        __bf16* sC = smem;            // 128 x 136 (pad) = 34816 B, fits
#pragma unroll
        for (int fa = 0; fa < 4; ++fa) {
#pragma unroll
            for (int fb = 0; fb < 2; ++fb) {
                const int rb  = wm * 64 + fa * 16 + (half << 3);
                const int col = wn * 32 + fb * 16 + lrow;
                const float bv = args.bias[n0 + col];
#pragma unroll
                for (int i = 0; i < 8; ++i) {
                    float v = acc[fa][fb][i] + bv;
                    if (MODE == 1) v = fmaxf(v, 0.0f);
                    sC[(rb + i) * 136 + col] = (__bf16)v;
                }
            }
        }
        __syncthreads();
#pragma unroll
        for (int it = 0; it < 8; ++it) {
            int c   = tid + it * 256;
            int row = c >> 4;
            int kc  = (c & 15) << 3;
            uint4 v = *reinterpret_cast<const uint4*>(&sC[row * 136 + kc]);
            *reinterpret_cast<uint4*>(
                args.Cbf + (size_t)(m0 + row) * args.ldc + n0 + kc) = v;
        }
    } else {
#pragma unroll
        for (int fa = 0; fa < 4; ++fa) {
#pragma unroll
            for (int fb = 0; fb < 2; ++fb) {
                const int rbase = m0 + wm * 64 + fa * 16 + (half << 3);
                const int col   = n0 + wn * 32 + fb * 16 + lrow;
                const float bv  = args.bias[col];
#pragma unroll
                for (int i = 0; i < 8; ++i) {
                    const int row = rbase + i;
                    float v = acc[fa][fb][i] + bv;
                    if (MODE == 2)
                        v += args.resid[(size_t)row * D_ + col];
                    args.Cf[(size_t)row * args.ldc + col] = v;
                }
            }
        }
    }
}

// ---------------- weight fp32 -> bf16 conversion ----------------
__global__ void convert_weights(const float* Wk, const float* Wv,
                                const float* bk, const float* bv,
                                const float* W1, const float* W2,
                                const float* Wo, const float* Wout,
                                __bf16* Wkvb, float* kvbias,
                                __bf16* W1b, __bf16* W2b,
                                __bf16* Wob, __bf16* Woutb) {
    const int stride = gridDim.x * blockDim.x;
    const int i0 = blockIdx.x * blockDim.x + threadIdx.x;
    for (int i = i0; i < D_ * D_; i += stride) {
        Wkvb[i]           = (__bf16)Wk[i];
        Wkvb[D_ * D_ + i] = (__bf16)Wv[i];
        Wob[i]            = (__bf16)Wo[i];
    }
    for (int i = i0; i < D_; i += stride) {
        kvbias[i]      = bk[i];
        kvbias[D_ + i] = bv[i];
    }
    for (int i = i0; i < FF_ * D_; i += stride) {
        W1b[i] = (__bf16)W1[i];
        W2b[i] = (__bf16)W2[i];
    }
    for (int i = i0; i < OUT_ * D_; i += stride) Woutb[i] = (__bf16)Wout[i];
}

// ---------------- q0 = (Wq @ cls + bq) / sqrt(dh), once ----------
__global__ void q0_kernel(const float* Wq, const float* bq, const float* cls,
                          float* q0s) {
    const int j = threadIdx.x;              // 0..511
    float acc = bq[j];
    const float* wr = Wq + (size_t)j * D_;
    for (int c = 0; c < D_; ++c) acc += cls[c] * wr[c];
    q0s[j] = acc * 0.125f;                  // 1/sqrt(64)
}

// ---------------- per-group attention (cls row) + Wo + residual + LN1 -----
__global__ __launch_bounds__(256) void attn_kernel(
    const __bf16* KV, const float* q0s, const float* cls, const float* bo,
    const __bf16* Wob, const float* ln1g, const float* ln1b,
    __bf16* h1b, float* h1f) {
    __shared__ float sc[H_][S_];
    __shared__ float red[H_][S_];
    __shared__ float qs[D_];
    __shared__ float ctxs[D_];
    __shared__ float uu[D_];

    const int g = blockIdx.x;
    const int t = threadIdx.x;              // 0..255

    qs[t]       = q0s[t];
    qs[t + 256] = q0s[t + 256];
    __syncthreads();

    // scores for cls query vs key row t, all 8 heads (b128 loads of K)
    const __bf16* krow = KV + ((size_t)g * S_ + t) * (2 * D_);
    float sv[H_];
    for (int h = 0; h < H_; ++h) {
        float a = 0.0f;
        const float*  qh = qs + h * DH_;
        const __bf16* kh = krow + h * DH_;
#pragma unroll
        for (int c8 = 0; c8 < 8; ++c8) {
            v8bf kv = *reinterpret_cast<const v8bf*>(kh + c8 * 8);
#pragma unroll
            for (int e = 0; e < 8; ++e)
                a += qh[c8 * 8 + e] * (float)kv[e];
        }
        sv[h] = a;
        sc[h][t]  = a;
        red[h][t] = a;
    }
    __syncthreads();

    // softmax over t per head
    for (int off = 128; off > 0; off >>= 1) {
        if (t < off)
#pragma unroll
            for (int h = 0; h < H_; ++h)
                red[h][t] = fmaxf(red[h][t], red[h][t + off]);
        __syncthreads();
    }
    float e[H_];
#pragma unroll
    for (int h = 0; h < H_; ++h) e[h] = __expf(sv[h] - red[h][0]);
    __syncthreads();
#pragma unroll
    for (int h = 0; h < H_; ++h) red[h][t] = e[h];
    __syncthreads();
    for (int off = 128; off > 0; off >>= 1) {
        if (t < off)
#pragma unroll
            for (int h = 0; h < H_; ++h) red[h][t] += red[h][t + off];
        __syncthreads();
    }
#pragma unroll
    for (int h = 0; h < H_; ++h) sc[h][t] = e[h] / red[h][0];
    __syncthreads();

    // ctx[d] = sum_t att[head(d)][t] * V[t][d]   (coalesced over d)
    {
        const __bf16* vbase = KV + (size_t)g * S_ * (2 * D_) + D_;
        const int d0 = t, d1 = t + 256;
        const int h0 = d0 >> 6, h1i = d1 >> 6;
        float a0 = 0.0f, a1 = 0.0f;
        for (int t2 = 0; t2 < S_; ++t2) {
            const __bf16* vr = vbase + (size_t)t2 * (2 * D_);
            a0 += sc[h0][t2]  * (float)vr[d0];
            a1 += sc[h1i][t2] * (float)vr[d1];
        }
        ctxs[d0] = a0;
        ctxs[d1] = a1;
    }
    __syncthreads();

    // u = cls + ctx @ Wo^T + bo     (b128 loads of Wo rows)
    for (int d = t; d < D_; d += 256) {
        float u = cls[d] + bo[d];
        const __bf16* wr = Wob + (size_t)d * D_;
        float a = 0.0f;
        for (int j8 = 0; j8 < 64; ++j8) {
            v8bf wv = *reinterpret_cast<const v8bf*>(wr + j8 * 8);
#pragma unroll
            for (int e = 0; e < 8; ++e)
                a += ctxs[j8 * 8 + e] * (float)wv[e];
        }
        uu[d] = u + a;
    }
    __syncthreads();

    // LayerNorm over 512
    float* r1 = &red[0][0];
    float* r2 = &red[2][0];
    r1[t] = uu[t] + uu[t + 256];
    r2[t] = uu[t] * uu[t] + uu[t + 256] * uu[t + 256];
    __syncthreads();
    for (int off = 128; off > 0; off >>= 1) {
        if (t < off) { r1[t] += r1[t + off]; r2[t] += r2[t + off]; }
        __syncthreads();
    }
    const float mean = r1[0] * (1.0f / D_);
    const float var  = r2[0] * (1.0f / D_) - mean * mean;
    const float rstd = rsqrtf(var + 1e-5f);
    for (int d = t; d < D_; d += 256) {
        float hv = (uu[d] - mean) * rstd * ln1g[d] + ln1b[d];
        h1f[(size_t)g * D_ + d] = hv;
        h1b[(size_t)g * D_ + d] = (__bf16)hv;
    }
}

// ---------------- LN2 over pre-LN rows -> bf16 ----------------
__global__ __launch_bounds__(256) void ln2_kernel(const float* preln,
                                                  const float* g2,
                                                  const float* b2,
                                                  __bf16* h2b) {
    __shared__ float r1[256];
    __shared__ float r2[256];
    const int g = blockIdx.x;
    const int t = threadIdx.x;
    const float a = preln[(size_t)g * D_ + t];
    const float b = preln[(size_t)g * D_ + t + 256];
    r1[t] = a + b;
    r2[t] = a * a + b * b;
    __syncthreads();
    for (int off = 128; off > 0; off >>= 1) {
        if (t < off) { r1[t] += r1[t + off]; r2[t] += r2[t + off]; }
        __syncthreads();
    }
    const float mean = r1[0] * (1.0f / D_);
    const float var  = r2[0] * (1.0f / D_) - mean * mean;
    const float rstd = rsqrtf(var + 1e-5f);
    h2b[(size_t)g * D_ + t]       = (__bf16)((a - mean) * rstd * g2[t] + b2[t]);
    h2b[(size_t)g * D_ + t + 256] =
        (__bf16)((b - mean) * rstd * g2[t + 256] + b2[t + 256]);
}

// ---------------------------------------------------------------------------
extern "C" void kernel_launch(void* const* d_in, const int* in_sizes, int n_in,
                              void* d_out, int out_size, void* d_ws,
                              size_t ws_size, hipStream_t stream) {
    const float* x    = (const float*)d_in[0];
    const float* cls  = (const float*)d_in[2];
    const float* Wq   = (const float*)d_in[3];
    const float* bq   = (const float*)d_in[4];
    const float* Wk   = (const float*)d_in[5];
    const float* bk   = (const float*)d_in[6];
    const float* Wv   = (const float*)d_in[7];
    const float* bv   = (const float*)d_in[8];
    const float* Wo   = (const float*)d_in[9];
    const float* bo   = (const float*)d_in[10];
    const float* ln1g = (const float*)d_in[11];
    const float* ln1b = (const float*)d_in[12];
    const float* W1   = (const float*)d_in[13];
    const float* b1   = (const float*)d_in[14];
    const float* W2   = (const float*)d_in[15];
    const float* b2   = (const float*)d_in[16];
    const float* ln2g = (const float*)d_in[17];
    const float* ln2b = (const float*)d_in[18];
    const float* Wout = (const float*)d_in[19];
    const float* bout = (const float*)d_in[20];

    char* w = (char*)d_ws;
    size_t o = 0;
    auto take = [&](size_t bytes) -> char* {
        o = (o + 255) & ~(size_t)255;
        char* p = w + o;
        o += bytes;
        return p;
    };
    __bf16* Wkvb   = (__bf16*)take((size_t)2 * D_ * D_ * 2);   // [1024][512]
    float*  kvbias = (float*) take((size_t)2 * D_ * 4);
    __bf16* W1b    = (__bf16*)take((size_t)FF_ * D_ * 2);
    __bf16* W2b    = (__bf16*)take((size_t)D_ * FF_ * 2);
    __bf16* Wob    = (__bf16*)take((size_t)D_ * D_ * 2);
    __bf16* Woutb  = (__bf16*)take((size_t)OUT_ * D_ * 2);
    float*  q0s    = (float*) take((size_t)D_ * 4);
    __bf16* KV     = (__bf16*)take((size_t)G_ * S_ * 2 * D_ * 2); // 128 MB
    __bf16* h1bb   = (__bf16*)take((size_t)G_ * D_ * 2);
    float*  h1f    = (float*) take((size_t)G_ * D_ * 4);
    __bf16* mid    = (__bf16*)take((size_t)G_ * FF_ * 2);
    float*  preln  = (float*) take((size_t)G_ * D_ * 4);
    __bf16* h2b    = (__bf16*)take((size_t)G_ * D_ * 2);

    convert_weights<<<256, 256, 0, stream>>>(Wk, Wv, bk, bv, W1, W2, Wo, Wout,
                                             Wkvb, kvbias, W1b, W2b, Wob, Woutb);
    q0_kernel<<<1, 512, 0, stream>>>(Wq, bq, cls, q0s);

    // K|V projection: M=65536, N=1024, K=512
    {
        GemmArgs a = {};
        a.xsrc = x; a.cls = cls; a.Bbf = Wkvb; a.bias = kvbias;
        a.Cbf = KV; a.K = D_; a.ldc = 2 * D_;
        gemm_wmma<0><<<dim3(8, 512), 256, 0, stream>>>(a);
    }

    attn_kernel<<<G_, 256, 0, stream>>>(KV, q0s, cls, bo, Wob, ln1g, ln1b,
                                        h1bb, h1f);

    // FFN1: relu(h1 @ W1^T + b1), M=256, N=2048, K=512
    {
        GemmArgs a = {};
        a.Abf = h1bb; a.Bbf = W1b; a.bias = b1; a.Cbf = mid;
        a.K = D_; a.ldc = FF_;
        gemm_wmma<1><<<dim3(16, 2), 256, 0, stream>>>(a);
    }

    // FFN2: mid @ W2^T + b2 + h1 -> preln, M=256, N=512, K=2048
    {
        GemmArgs a = {};
        a.Abf = mid; a.Bbf = W2b; a.bias = b2; a.resid = h1f; a.Cf = preln;
        a.K = FF_; a.ldc = D_;
        gemm_wmma<2><<<dim3(4, 2), 256, 0, stream>>>(a);
    }

    ln2_kernel<<<G_, 256, 0, stream>>>(preln, ln2g, ln2b, h2b);

    // out = h2 @ Wout^T + bout, M=256, N=128, K=512 -> d_out (f32)
    {
        GemmArgs a = {};
        a.Abf = h2b; a.Bbf = Woutb; a.bias = bout; a.Cf = (float*)d_out;
        a.K = D_; a.ldc = OUT_;
        gemm_wmma<3><<<dim3(1, 2), 256, 0, stream>>>(a);
    }
}
